// BinaryTreeLSTM_75943611728148
// MI455X (gfx1250) — matrix operand order repaired and checked by reading, not compiled, not run on hardware
//
#include <hip/hip_runtime.h>
#include <hip/hip_bf16.h>
#include <math.h>

// Problem constants (match reference)
#define Bb 256
#define Ll 32
#define Dd 512
#define Hh 512
#define NEG_INF_F (-1000000000.0f)
#define LOGIT_SCALE 0.04419417382415922f  // 1/sqrt(512)

typedef unsigned short u16;
typedef unsigned int   u32;
typedef __attribute__((ext_vector_type(8)))  u16    u16x8;
typedef __attribute__((ext_vector_type(16))) __bf16 bf16x16;
typedef __attribute__((ext_vector_type(8)))  float  v8f;

// round-to-nearest-even f32 -> bf16 (bit pattern as u16)
__device__ __forceinline__ u16 f2bf(float f) {
  u32 u = __float_as_uint(f);
  u32 r = u + 0x7FFFu + ((u >> 16) & 1u);
  return (u16)(r >> 16);
}

// Build a 16-element bf16 fragment from two contiguous 8-element (16B) chunks.
__device__ __forceinline__ bf16x16 frag16(const u16* p0, const u16* p1) {
  union { struct { u16x8 lo; u16x8 hi; } p; bf16x16 f; } u;
  u.p.lo = *(const u16x8*)p0;   // global_load_b128
  u.p.hi = *(const u16x8*)p1;   // global_load_b128
  return u.f;
}

__device__ __forceinline__ float sigm(float x) { return 1.0f / (1.0f + __expf(-x)); }

// ---------------------------------------------------------------------------
// elementwise f32 -> bf16 conversion
__global__ __launch_bounds__(256) void cvt_bf16_k(const float* __restrict__ s,
                                                  u16* __restrict__ d, int n) {
  int i = blockIdx.x * 256 + threadIdx.x;
  if (i < n) d[i] = f2bf(s[i]);
}

__global__ __launch_bounds__(256) void zero_f32_k(float* __restrict__ p, int n) {
  int i = blockIdx.x * 256 + threadIdx.x;
  if (i < n) p[i] = 0.0f;
}

// ---------------------------------------------------------------------------
// Word embedding GEMM: hc[r, n] = sum_d x[r,d]*W_word[n,d] + b_word[n]
// r in [0, B*L), n in [0, 2H). n<H -> h (f32 + bf16 mirror), else c (f32).
// grid = (M/16, N/(4*64)); block = 128 (4 waves); wave tile = 16x64 (4 accs).
// K-loop FULLY UNROLLED (16 stages): pure SSA, no loop-carried fragment
// registers -> no rotation copies; scheduler hoists b128 loads freely.
__global__ __launch_bounds__(128, 1)
void word_gemm_k(const u16* __restrict__ X, const u16* __restrict__ W,
                 const float* __restrict__ bias,
                 float* __restrict__ h, float* __restrict__ c,
                 u16* __restrict__ hbf) {
  const int lane = threadIdx.x & 31;
  const int wave = threadIdx.x >> 5;
  const int m0   = blockIdx.x * 16;
  const int n0   = (blockIdx.y * 4 + wave) * 64;
  const int half = lane >> 4;
  const int l15  = lane & 15;

  const u16* arow  = X + (size_t)(m0 + l15) * Dd;
  const u16* wbase = W + (size_t)(n0 + l15) * Dd + half * 16;  // +g*16*Dd per acc

  v8f acc[4] = {v8f{}, v8f{}, v8f{}, v8f{}};
#pragma unroll
  for (int kb = 0; kb < Dd; kb += 32) {
    bf16x16 a = frag16(arow + kb + half * 16, arow + kb + half * 16 + 8);
#pragma unroll
    for (int g = 0; g < 4; ++g) {
      const u16* p = wbase + g * 16 * Dd + kb;   // constant imm offsets
      bf16x16 bf = frag16(p, p + 8);
      acc[g] = __builtin_amdgcn_wmma_f32_16x16x32_bf16(
          false, a, false, bf, (short)0, acc[g], false, false);
    }
  }

#pragma unroll
  for (int g = 0; g < 4; ++g) {
    const int col = n0 + g * 16 + l15;
    const float bv = bias[col];
#pragma unroll
    for (int e = 0; e < 8; ++e) {
      int row = m0 + e + half * 8;                // row == b*L + l exactly
      float v = acc[g][e] + bv;
      if (col < Hh) {
        size_t o = (size_t)row * Hh + col;
        h[o] = v;
        hbf[o] = f2bf(v);
      } else {
        c[(size_t)row * Hh + (col - Hh)] = v;
      }
    }
  }
}

// ---------------------------------------------------------------------------
// Compose + gates + logit dot, fused. One wave owns 16 rows x 16 h-columns and
// computes all 5 gate pre-activations in registers (5 accumulators, A reused
// 5x per K-step). A row r = b*n + j; concat(h[b,j], h[b,j+1]) is CONTIGUOUS in
// the (B,L,H) bf16 buffer, so the A operand is a flat 1024-element run.
// K-loop FULLY UNROLLED (32 stages, 160 WMMAs): pure SSA, no copies; all
// weight addresses are wbase + 24-bit immediates (max 4MB < 8MB reach).
// grid = (B*n/16, H/64); block = 128 (4 waves, one 16-col group each).
__global__ __launch_bounds__(128, 1)
void compose_gates_k(const u16* __restrict__ hbf, const float* __restrict__ cbuf,
                     const u16* __restrict__ Wc, const float* __restrict__ bias,
                     const float* __restrict__ q,
                     float* __restrict__ newh, float* __restrict__ newc,
                     float* __restrict__ logits, int n) {
  const int lane = threadIdx.x & 31;
  const int wave = threadIdx.x >> 5;
  const int m0   = blockIdx.x * 16;
  const int col0 = (blockIdx.y * 4 + wave) * 16;
  const int half = lane >> 4;
  const int l15  = lane & 15;
  const int K    = 2 * Hh;

  // A row pointer for this lane's row
  int row = m0 + l15;
  int b   = row / n;
  int j   = row - b * n;
  const u16* arow  = hbf + (size_t)(b * Ll + j) * Hh;        // 1024 contiguous bf16
  const u16* wbase = Wc + (size_t)(col0 + l15) * K + half * 16;  // +g*Hh*K per gate

  v8f acc[5] = {v8f{}, v8f{}, v8f{}, v8f{}, v8f{}};
#pragma unroll
  for (int kb = 0; kb < 2 * Hh; kb += 32) {
    bf16x16 a = frag16(arow + kb + half * 16, arow + kb + half * 16 + 8);
#pragma unroll
    for (int g = 0; g < 5; ++g) {
      const u16* p = wbase + g * Hh * K + kb;    // constant imm offsets
      bf16x16 bf = frag16(p, p + 8);
      acc[g] = __builtin_amdgcn_wmma_f32_16x16x32_bf16(
          false, a, false, bf, (short)0, acc[g], false, false);
    }
  }

  // Epilogue: TreeLSTM gates, entirely from accumulator registers.
  const int col = col0 + l15;
  const float bi = bias[col];
  const float bl = bias[Hh + col] + 1.0f;     // forget bias +1 folded in
  const float br = bias[2 * Hh + col] + 1.0f;
  const float bu = bias[3 * Hh + col];
  const float bo = bias[4 * Hh + col];
  const float qc = q[col];

#pragma unroll
  for (int e = 0; e < 8; ++e) {
    int r  = m0 + e + half * 8;
    int bb = r / n;
    int jj = r - bb * n;
    size_t crow = (size_t)(bb * Ll + jj) * Hh + col;
    float cl = cbuf[crow];
    float cr = cbuf[crow + Hh];               // c[b, j+1] is the next row

    float cn = cl * sigm(acc[1][e] + bl) + cr * sigm(acc[2][e] + br) +
               tanhf(acc[3][e] + bu) * sigm(acc[0][e] + bi);
    float hn = sigm(acc[4][e] + bo) * tanhf(cn);

    size_t orow = (size_t)r * Hh + col;
    newh[orow] = hn;
    newc[orow] = cn;

    // partial logit: reduce q*h over the 16 lanes sharing this row
    float lp = qc * hn;
    lp += __shfl_xor(lp, 8, 16);
    lp += __shfl_xor(lp, 4, 16);
    lp += __shfl_xor(lp, 2, 16);
    lp += __shfl_xor(lp, 1, 16);
    if (l15 == 0) atomicAdd(&logits[r], lp * LOGIT_SCALE);
  }
}

// ---------------------------------------------------------------------------
// Per-batch masked argmax (first-max tiebreak like jnp.argmax). 1 wave per b.
__global__ __launch_bounds__(32)
void argmax_k(const float* __restrict__ logits, const int* __restrict__ length,
              int n, int step, int* __restrict__ amax) {
  int b = blockIdx.x;
  int lane = threadIdx.x;
  float best = -3.0e38f;
  int bj = 0;
  for (int jx = lane; jx < n; jx += 32) {
    float v = ((step + 1 + jx) < length[b]) ? logits[b * n + jx] : NEG_INF_F;
    if (v > best) { best = v; bj = jx; }
  }
#pragma unroll
  for (int s = 16; s >= 1; s >>= 1) {
    float ov = __shfl_xor(best, s, 32);
    int   oj = __shfl_xor(bj, s, 32);
    if (ov > best || (ov == best && oj < bj)) { best = ov; bj = oj; }
  }
  if (lane == 0) amax[b] = bj;
}

// ---------------------------------------------------------------------------
// Merge: j==amax -> composed; j<amax -> keep h[b,j]; j>amax -> keep h[b,j+1].
// If (step+1) >= length[b] the step is a no-op truncate. Writes ping-pong
// h/c f32 + bf16 mirror of h for the next step's WMMA A-operand.
__global__ __launch_bounds__(256)
void merge_k(const float* __restrict__ hin, const float* __restrict__ cin,
             const float* __restrict__ newh, const float* __restrict__ newc,
             const int* __restrict__ amax, const int* __restrict__ length,
             float* __restrict__ hout, float* __restrict__ cout,
             u16* __restrict__ hbfout, int n, int step) {
  int idx = blockIdx.x * 256 + threadIdx.x;
  int total = Bb * n * Hh;
  if (idx >= total) return;
  int col = idx & (Hh - 1);
  int rj  = idx >> 9;           // /H
  int b   = rj / n;
  int j   = rj - b * n;
  size_t inrow = (size_t)(b * Ll + j) * Hh + col;
  float hv, cv;
  bool done = (step + 1) < length[b];
  if (!done) {
    hv = hin[inrow]; cv = cin[inrow];
  } else {
    int am = amax[b];
    if (j == am) {
      size_t nr = (size_t)rj * Hh + col;
      hv = newh[nr]; cv = newc[nr];
    } else if (j < am) {
      hv = hin[inrow]; cv = cin[inrow];
    } else {
      hv = hin[inrow + Hh]; cv = cin[inrow + Hh];
    }
  }
  size_t orow = (size_t)(b * Ll + j) * Hh + col;
  hout[orow] = hv;
  cout[orow] = cv;
  hbfout[orow] = f2bf(hv);
}

__global__ __launch_bounds__(256)
void copyout_k(const float* __restrict__ hin, float* __restrict__ out) {
  int idx = blockIdx.x * 256 + threadIdx.x;
  if (idx >= Bb * Hh) return;
  int b = idx >> 9, k = idx & (Hh - 1);
  out[idx] = hin[(size_t)(b * Ll) * Hh + k];
}

// ---------------------------------------------------------------------------
extern "C" void kernel_launch(void* const* d_in, const int* in_sizes, int n_in,
                              void* d_out, int out_size, void* d_ws, size_t ws_size,
                              hipStream_t stream) {
  const float* x     = (const float*)d_in[0];   // (B,L,D)
  const int*   len   = (const int*)d_in[1];     // (B,)
  const float* Ww    = (const float*)d_in[2];   // (2H,D)
  const float* bw    = (const float*)d_in[3];   // (2H,)
  const float* Wc    = (const float*)d_in[4];   // (5H,2H)
  const float* bc    = (const float*)d_in[5];   // (5H,)
  const float* q     = (const float*)d_in[6];   // (H,)
  float* out = (float*)d_out;
  (void)in_sizes; (void)n_in; (void)out_size; (void)ws_size;

  // Workspace carve-up (256B aligned slices)
  size_t off = 0;
  auto carve = [&](size_t bytes) -> void* {
    void* p = (char*)d_ws + off;
    off += (bytes + 255) & ~(size_t)255;
    return p;
  };
  const size_t HC = (size_t)Bb * Ll * Hh;       // h/c element count
  float* h0  = (float*)carve(HC * 4);
  float* h1  = (float*)carve(HC * 4);
  float* c0  = (float*)carve(HC * 4);
  float* c1  = (float*)carve(HC * 4);
  u16*   hb0 = (u16*)carve(HC * 2);
  u16*   hb1 = (u16*)carve(HC * 2);
  u16*   xbf = (u16*)carve((size_t)Bb * Ll * Dd * 2);
  u16*   Wwb = (u16*)carve((size_t)2 * Hh * Dd * 2);
  u16*   Wcb = (u16*)carve((size_t)5 * Hh * 2 * Hh * 2);
  float* nh  = (float*)carve((size_t)Bb * (Ll - 1) * Hh * 4);
  float* nc  = (float*)carve((size_t)Bb * (Ll - 1) * Hh * 4);
  float* lg  = (float*)carve((size_t)Bb * (Ll - 1) * 4);
  int*   am  = (int*)carve((size_t)Bb * 4);

  // One-time f32 -> bf16 conversions (weights are L2-resident thereafter)
  {
    int nx = Bb * Ll * Dd;
    cvt_bf16_k<<<(nx + 255) / 256, 256, 0, stream>>>(x, xbf, nx);
    int nw = 2 * Hh * Dd;
    cvt_bf16_k<<<(nw + 255) / 256, 256, 0, stream>>>(Ww, Wwb, nw);
    int ncmp = 5 * Hh * 2 * Hh;
    cvt_bf16_k<<<(ncmp + 255) / 256, 256, 0, stream>>>(Wc, Wcb, ncmp);
  }

  // Word embedding GEMM: (8192 x 512) x (512 x 1024)
  {
    dim3 g((Bb * Ll) / 16, (2 * Hh) / 256);     // (512, 4)
    word_gemm_k<<<g, 128, 0, stream>>>(xbf, Wwb, bw, h0, c0, hb0);
  }

  float *hcur = h0, *ccur = c0, *hnxt = h1, *cnxt = c1;
  u16 *hbcur = hb0, *hbnxt = hb1;

  for (int i = 0; i < Ll - 1; ++i) {
    int n = Ll - 1 - i;
    int M = Bb * n;                              // multiple of 16

    zero_f32_k<<<(Bb * n + 255) / 256, 256, 0, stream>>>(lg, Bb * n);

    dim3 gc(M / 16, Hh / 64);                    // (16n, 8)
    compose_gates_k<<<gc, 128, 0, stream>>>(hbcur, ccur, Wcb, bc, q, nh, nc, lg, n);

    argmax_k<<<Bb, 32, 0, stream>>>(lg, len, n, i, am);

    int total = Bb * n * Hh;
    merge_k<<<(total + 255) / 256, 256, 0, stream>>>(
        hcur, ccur, nh, nc, am, len, hnxt, cnxt, hbnxt, n, i);

    { float* t = hcur; hcur = hnxt; hnxt = t; }
    { float* t = ccur; ccur = cnxt; cnxt = t; }
    { u16* t = hbcur; hbcur = hbnxt; hbnxt = t; }
  }

  copyout_k<<<(Bb * Hh + 255) / 256, 256, 0, stream>>>(hcur, out);
}
